// Attention_23914377904826
// MI455X (gfx1250) — matrix operand order repaired and checked by reading, not compiled, not run on hardware
//
#include <hip/hip_runtime.h>
#include <hip/hip_bf16.h>

#define DIM_   2048
#define SLEN   2048
#define BATCH  2
#define NH     32
#define NKV    8
#define HD     64
#define MASKV  -32768.0f

typedef _Float16 h16;
typedef __attribute__((ext_vector_type(16))) _Float16 v16h;
typedef __attribute__((ext_vector_type(8)))  _Float16 v8h;
typedef __attribute__((ext_vector_type(8)))  float    v8f;

static __device__ __forceinline__ v8f wmma_f16(v16h a, v16h b, v8f c) {
  // D = A(16x32) * B(32x16) + C, f32 accumulate
  return __builtin_amdgcn_wmma_f32_16x16x32_f16(false, a, false, b, (short)0, c, false, false);
}

// CDNA5 async global->LDS copy (ASYNCcnt-tracked). lds_off is the byte offset
// inside the workgroup's LDS allocation (our only __shared__ array => offset 0).
static __device__ __forceinline__ void async_load_b128(unsigned lds_off, const h16* g) {
  asm volatile("global_load_async_to_lds_b128 %0, %1, off"
               :: "v"(lds_off), "v"((unsigned long long)(uintptr_t)g)
               : "memory");
}
static __device__ __forceinline__ void wait_async0() {
  asm volatile("s_wait_asynccnt 0" ::: "memory");
}

// DPP16 butterfly reductions across each 16-lane group (VALU pipe, no LDS).
template<int CTRL>
static __device__ __forceinline__ float dpp_f32(float x) {
  return __int_as_float(
      __builtin_amdgcn_update_dpp(0, __float_as_int(x), CTRL, 0xf, 0xf, true));
}
static __device__ __forceinline__ float row16_max(float x) {
  x = fmaxf(x, dpp_f32<0xB1>(x));    // quad_perm(1,0,3,2)
  x = fmaxf(x, dpp_f32<0x4E>(x));    // quad_perm(2,3,0,1)
  x = fmaxf(x, dpp_f32<0x141>(x));   // row_half_mirror
  x = fmaxf(x, dpp_f32<0x140>(x));   // row_mirror
  return x;
}
static __device__ __forceinline__ float row16_sum(float x) {
  x += dpp_f32<0xB1>(x);
  x += dpp_f32<0x4E>(x);
  x += dpp_f32<0x141>(x);
  x += dpp_f32<0x140>(x);
  return x;
}

// A-fragment (16x32 f16, MxK): lane&15 = row; lane-half picks K sub-chunk:
// lanes 0-15 hold K {0..7,16..23}, lanes 16-31 hold K {8..15,24..31}.
static __device__ __forceinline__ v16h load_a_frag(const h16* base, int ld, int lane) {
  const int hf = (lane >> 4) & 1;
  const int r  = lane & 15;
  const h16* p = base + (size_t)r * ld + hf * 8;
  v8h lo = *reinterpret_cast<const v8h*>(p);
  v8h hi = *reinterpret_cast<const v8h*>(p + 16);
  return __builtin_shufflevector(lo, hi, 0,1,2,3,4,5,6,7,8,9,10,11,12,13,14,15);
}

// B-fragment (32x16 f16, KxN) where B[k][n] = W[n][k]: lane&15 = n (row of W),
// lanes 0-15 hold K {0..7,16..23}, lanes 16-31 hold K {8..15,24..31}.
static __device__ __forceinline__ v16h load_b_frag(const h16* base, int ld, int lane) {
  const int hf = (lane >> 4) & 1;
  const int n  = lane & 15;
  const h16* p = base + (size_t)n * ld + hf * 8;
  v8h lo = *reinterpret_cast<const v8h*>(p);
  v8h hi = *reinterpret_cast<const v8h*>(p + 16);
  return __builtin_shufflevector(lo, hi, 0,1,2,3,4,5,6,7,8,9,10,11,12,13,14,15);
}

__global__ void f32_to_f16(const float* __restrict__ in, h16* __restrict__ out, int n) {
  int i = blockIdx.x * blockDim.x + threadIdx.x;
  if (i < n) out[i] = (h16)in[i];
}

// C[M,N] = A[M,K] @ W[N,K]^T.
// Workgroup = 4 waves (128 threads), workgroup tile 64(M) x 256(N); wave wj owns
// the 64x64 C tile at n0 + 64*wj. The shared 64x32 A slab is staged in LDS via
// async global->LDS copies, double-buffered so the DMA for k+32 overlaps the
// 16 WMMAs at k. W fragments stream directly from global (L2-resident).
template<int OUT_F32>
__global__ void __launch_bounds__(128, 1)
gemm_xwT(const h16* __restrict__ A, const h16* __restrict__ W,
         void* __restrict__ Cp, int M, int N, int K) {
  __shared__ __align__(16) h16 atile[2][64 * 32];   // 2 x 4KB
  const int tid  = threadIdx.x;                     // 0..127
  const int lane = tid & 31;
  const int wj   = tid >> 5;                        // wave index 0..3
  const int tilesN = N >> 8;
  const int m0 = (blockIdx.x / tilesN) << 6;
  const int n0 = ((blockIdx.x % tilesN) << 8) + (wj << 6);

  const v8f zf = {0.f,0.f,0.f,0.f,0.f,0.f,0.f,0.f};
  v8f acc[4][4];
#pragma unroll
  for (int i = 0; i < 4; i++)
#pragma unroll
    for (int j = 0; j < 4; j++) acc[i][j] = zf;

  // copy the 64x32 A slab at column k0 into LDS buffer `buf` (2 x 16B per thread)
  auto issue_copy = [&](int k0, int buf) {
#pragma unroll
    for (int c = 0; c < 2; c++) {
      const int chunk = tid * 2 + c;        // 0..255
      const int row  = chunk >> 2;          // 0..63
      const int quad = chunk & 3;           // 16B quad within the 64B row
      const unsigned lds_off = (unsigned)(buf * (64 * 32 * 2) + row * 64 + quad * 16);
      async_load_b128(lds_off, A + (size_t)(m0 + row) * K + k0 + quad * 8);
    }
  };

  issue_copy(0, 0);
  const int nk = K >> 5;
  for (int ik = 0; ik < nk; ik++) {
    const int k0  = ik << 5;
    const int buf = ik & 1;
    wait_async0();          // our async writes into `buf` are done
    __syncthreads();        // everyone's are; previous reads of buf^1 retired
    if (ik + 1 < nk) issue_copy(k0 + 32, buf ^ 1);

    v16h a[4], b[4];
#pragma unroll
    for (int i = 0; i < 4; i++)
      a[i] = load_a_frag(&atile[buf][(16 * i) * 32], 32, lane);
#pragma unroll
    for (int j = 0; j < 4; j++)
      b[j] = load_b_frag(W + (size_t)(n0 + 16 * j) * K + k0, K, lane);
    if (ik + 1 < nk)
      __builtin_prefetch(W + (size_t)(n0 + (lane & 15)) * K + k0 + 32, 0, 3);
#pragma unroll
    for (int i = 0; i < 4; i++)
#pragma unroll
      for (int j = 0; j < 4; j++)
        acc[i][j] = wmma_f16(a[i], b[j], acc[i][j]);
  }

  const int col = lane & 15;
  const int rh  = (lane >> 4) & 1;
#pragma unroll
  for (int i = 0; i < 4; i++)
#pragma unroll
    for (int j = 0; j < 4; j++)
#pragma unroll
      for (int v = 0; v < 8; v++) {
        const int m = m0 + 16 * i + v + 8 * rh;   // C layout: VGPR v -> row v + 8*half
        const int n = n0 + 16 * j + col;
        const float val = acc[i][j][v];
        if (OUT_F32) ((float*)Cp)[(size_t)m * N + n] = val;
        else         ((h16*)Cp)[(size_t)m * N + n] = (h16)val;
      }
}

// RoPE on Q in place, folding 1/sqrt(HD)=0.125 into Q for the score GEMM.
__global__ void rope_q_scale(h16* __restrict__ Q, const float* __restrict__ fc,
                             const float* __restrict__ fs) {
  int i = blockIdx.x * blockDim.x + threadIdx.x;  // B*S*NH*(HD/2)
  int j = i & 31;
  int h = (i >> 5) & 31;
  int s = (i >> 10) & 2047;
  int b = i >> 21;
  float c = fc[s * 32 + j], sn = fs[s * 32 + j];
  h16* p = Q + ((size_t)(b * SLEN + s)) * DIM_ + h * HD + 2 * j;
  float tr = (float)p[0], ti = (float)p[1];
  p[0] = (h16)((tr * c - ti * sn) * 0.125f);
  p[1] = (h16)((tr * sn + ti * c) * 0.125f);
}

// RoPE on K + relayout [b,s,kvh,d] -> [b,kvh,s,d]
__global__ void rope_k_relayout(const h16* __restrict__ Kraw, h16* __restrict__ Kb,
                                const float* __restrict__ fc, const float* __restrict__ fs) {
  int i = blockIdx.x * blockDim.x + threadIdx.x;  // B*S*NKV*(HD/2)
  int j = i & 31;
  int kvh = (i >> 5) & 7;
  int s = (i >> 8) & 2047;
  int b = i >> 19;
  float c = fc[s * 32 + j], sn = fs[s * 32 + j];
  const h16* src = Kraw + ((size_t)(b * SLEN + s)) * (NKV * HD) + kvh * HD + 2 * j;
  h16* dst = Kb + (((size_t)(b * NKV + kvh)) * SLEN + s) * HD + 2 * j;
  float tr = (float)src[0], ti = (float)src[1];
  dst[0] = (h16)(tr * c - ti * sn);
  dst[1] = (h16)(tr * sn + ti * c);
}

// V [b,s,kvh,d] -> Vt [b,kvh,d,s] so PV B-fragments are contiguous along s.
__global__ void v_transpose(const h16* __restrict__ Vraw, h16* __restrict__ Vt) {
  int i = blockIdx.x * blockDim.x + threadIdx.x;  // B*S*NKV*HD
  int d = i & 63;
  int kvh = (i >> 6) & 7;
  int s = (i >> 9) & 2047;
  int b = i >> 20;
  Vt[(((size_t)(b * NKV + kvh)) * HD + d) * SLEN + s] =
      Vraw[((size_t)(b * SLEN + s)) * (NKV * HD) + kvh * HD + d];
}

// Flash attention: one wave per (b, h, 16-query tile). Online softmax, causal.
// Only the final key block can touch the causal diagonal => interior blocks
// skip the mask ALU entirely.
__global__ void __launch_bounds__(32, 1)
flash_attn(const h16* __restrict__ Q, const h16* __restrict__ Kb,
           const h16* __restrict__ Vt, h16* __restrict__ Ob) {
  __shared__ __align__(32) h16 pl[16 * 32];   // P tile staging (C-layout -> A-layout)
  const int lane = threadIdx.x & 31;
  const int qt = blockIdx.x;                  // query tile (16 rows)
  const int bh = blockIdx.y;
  const int h = bh & (NH - 1);
  const int b = bh >> 5;
  const int kvh = h >> 2;                     // N_REP = 4
  const int col = lane & 15;
  const int rh  = (lane >> 4) & 1;

  const h16* qbase = Q  + ((size_t)(b * SLEN + qt * 16)) * DIM_ + h * HD;
  const h16* kbase = Kb + ((size_t)(b * NKV + kvh)) * SLEN * HD;
  const h16* vbase = Vt + ((size_t)(b * NKV + kvh)) * HD * SLEN;

  v16h qf0 = load_a_frag(qbase,      DIM_, lane);   // d 0..31
  v16h qf1 = load_a_frag(qbase + 32, DIM_, lane);   // d 32..63

  const v8f zf = {0.f,0.f,0.f,0.f,0.f,0.f,0.f,0.f};
  v8f o0 = zf, o1 = zf, o2 = zf, o3 = zf;
  float mrow[8], lrow[8];
#pragma unroll
  for (int v = 0; v < 8; v++) { mrow[v] = -1e30f; lrow[v] = 0.f; }

  auto process_block = [&](int kb, bool masked) {
    const h16* kp = kbase + (size_t)kb * HD;
    v16h k00 = load_b_frag(kp,                HD, lane);  // keys kb..kb+15, d 0..31
    v16h k01 = load_b_frag(kp + 32,           HD, lane);  // d 32..63
    v16h k10 = load_b_frag(kp + 16 * HD,      HD, lane);  // keys kb+16..kb+31
    v16h k11 = load_b_frag(kp + 16 * HD + 32, HD, lane);
    // V fragments are independent of the scores: issue early so the softmax
    // ALU work hides their latency.
    v16h vf0 = load_b_frag(vbase + kb,                     SLEN, lane);
    v16h vf1 = load_b_frag(vbase + (size_t)16 * SLEN + kb, SLEN, lane);
    v16h vf2 = load_b_frag(vbase + (size_t)32 * SLEN + kb, SLEN, lane);
    v16h vf3 = load_b_frag(vbase + (size_t)48 * SLEN + kb, SLEN, lane);
    if (!masked)
      __builtin_prefetch(kp + (size_t)32 * HD + lane * 8, 0, 3);

    v8f s0 = zf, s1 = zf;
    s0 = wmma_f16(qf0, k00, s0);  s0 = wmma_f16(qf1, k01, s0);
    s1 = wmma_f16(qf0, k10, s1);  s1 = wmma_f16(qf1, k11, s1);

    const int key0 = kb + col, key1 = kb + 16 + col;
#pragma unroll
    for (int v = 0; v < 8; v++) {
      float x0 = s0[v];
      float x1 = s1[v];
      if (masked) {
        const int qg = qt * 16 + v + 8 * rh;
        x0 += (key0 > qg ? MASKV : 0.0f);
        x1 += (key1 > qg ? MASKV : 0.0f);
      }
      const float mx = row16_max(fmaxf(x0, x1));
      const float mnew = fmaxf(mrow[v], mx);
      const float corr = __expf(mrow[v] - mnew);
      const float p0 = __expf(x0 - mnew);
      const float p1 = __expf(x1 - mnew);
      const float rs = row16_sum(p0 + p1);
      lrow[v] = lrow[v] * corr + rs;
      mrow[v] = mnew;
      o0[v] *= corr; o1[v] *= corr; o2[v] *= corr; o3[v] *= corr;
      const int r = v + 8 * rh;
      pl[r * 32 + col]      = (h16)p0;
      pl[r * 32 + 16 + col] = (h16)p1;
    }
    __syncthreads();
    v16h pa = load_a_frag(pl, 32, lane);     // P 16x32 in A layout
    __syncthreads();

    o0 = wmma_f16(pa, vf0, o0);
    o1 = wmma_f16(pa, vf1, o1);
    o2 = wmma_f16(pa, vf2, o2);
    o3 = wmma_f16(pa, vf3, o3);
  };

  const int nkb = (qt >> 1) + 1;   // ceil((qt+1)*16 / 32) key blocks
  for (int ib = 0; ib < nkb - 1; ib++) process_block(ib * 32, false);
  process_block((nkb - 1) * 32, true);   // only this block sees the diagonal

  h16* obase = Ob + ((size_t)(b * SLEN + qt * 16)) * DIM_ + h * HD;
#pragma unroll
  for (int v = 0; v < 8; v++) {
    const float inv = 1.0f / lrow[v];
    const int r = v + 8 * rh;
    obase[(size_t)r * DIM_ +      col] = (h16)(o0[v] * inv);
    obase[(size_t)r * DIM_ + 16 + col] = (h16)(o1[v] * inv);
    obase[(size_t)r * DIM_ + 32 + col] = (h16)(o2[v] * inv);
    obase[(size_t)r * DIM_ + 48 + col] = (h16)(o3[v] * inv);
  }
}

extern "C" void kernel_launch(void* const* d_in, const int* in_sizes, int n_in,
                              void* d_out, int out_size, void* d_ws, size_t ws_size,
                              hipStream_t stream) {
  (void)in_sizes; (void)n_in; (void)out_size; (void)ws_size;
  const float* x  = (const float*)d_in[0];
  const float* fc = (const float*)d_in[1];
  const float* fs = (const float*)d_in[2];
  const float* wq = (const float*)d_in[3];
  const float* wk = (const float*)d_in[4];
  const float* wv = (const float*)d_in[5];
  const float* wo = (const float*)d_in[6];
  float* out = (float*)d_out;

  char* ws = (char*)d_ws;
  size_t off = 0;
  auto alloc = [&](size_t elems) {
    h16* p = (h16*)(ws + off);
    off += ((elems * sizeof(h16)) + 255) & ~(size_t)255;
    return p;
  };
  const size_t M = (size_t)BATCH * SLEN;          // 4096
  h16* xb   = alloc(M * DIM_);                    // 16 MB
  h16* wqb  = alloc((size_t)DIM_ * DIM_);         //  8 MB
  h16* wkb  = alloc((size_t)NKV * HD * DIM_);     //  2 MB
  h16* wvb  = alloc((size_t)NKV * HD * DIM_);     //  2 MB
  h16* wob  = alloc((size_t)DIM_ * DIM_);         //  8 MB
  h16* Qb   = alloc(M * DIM_);                    // 16 MB
  h16* Kraw = alloc(M * NKV * HD);                //  4 MB
  h16* Vraw = alloc(M * NKV * HD);                //  4 MB
  h16* Kb   = alloc(M * NKV * HD);                //  4 MB
  h16* Vt   = alloc(M * NKV * HD);                //  4 MB
  h16* Ab   = alloc(M * DIM_);                    // 16 MB

  auto cvt = [&](const float* src, h16* dst, size_t n) {
    f32_to_f16<<<(unsigned)((n + 255) / 256), 256, 0, stream>>>(src, dst, (int)n);
  };
  cvt(x,  xb,  M * DIM_);
  cvt(wq, wqb, (size_t)DIM_ * DIM_);
  cvt(wk, wkb, (size_t)NKV * HD * DIM_);
  cvt(wv, wvb, (size_t)NKV * HD * DIM_);
  cvt(wo, wob, (size_t)DIM_ * DIM_);

  // QKV projections (f16 in/out, f32 accumulate); 128 threads = 4 waves / block
  gemm_xwT<0><<<(unsigned)((M / 64) * (DIM_ / 256)),       128, 0, stream>>>(xb, wqb, Qb,   (int)M, DIM_,     DIM_);
  gemm_xwT<0><<<(unsigned)((M / 64) * ((NKV * HD) / 256)), 128, 0, stream>>>(xb, wkb, Kraw, (int)M, NKV * HD, DIM_);
  gemm_xwT<0><<<(unsigned)((M / 64) * ((NKV * HD) / 256)), 128, 0, stream>>>(xb, wvb, Vraw, (int)M, NKV * HD, DIM_);

  rope_q_scale   <<<(BATCH * SLEN * NH * (HD / 2)) / 256,  256, 0, stream>>>(Qb, fc, fs);
  rope_k_relayout<<<(BATCH * SLEN * NKV * (HD / 2)) / 256, 256, 0, stream>>>(Kraw, Kb, fc, fs);
  v_transpose    <<<(BATCH * SLEN * NKV * HD) / 256,       256, 0, stream>>>(Vraw, Vt);

  dim3 fgrid(SLEN / 16, BATCH * NH);
  flash_attn<<<fgrid, 32, 0, stream>>>(Qb, Kb, Vt, Ab);

  // Output projection, f32 epilogue into d_out
  gemm_xwT<1><<<(unsigned)((M / 64) * (DIM_ / 256)), 128, 0, stream>>>(Ab, wob, out, (int)M, DIM_, DIM_);
}
